// GatedMLA_42296837931251
// MI455X (gfx1250) — matrix operand order repaired and checked by reading, not compiled, not run on hardware
//
#include <hip/hip_runtime.h>
#include <hip/hip_bf16.h>

typedef __bf16 bf16_t;
typedef __attribute__((ext_vector_type(16))) __bf16 v16bf;
typedef __attribute__((ext_vector_type(8)))  float  v8f;
typedef unsigned int u32;
typedef __attribute__((ext_vector_type(4))) u32 u32x4;
typedef __attribute__((ext_vector_type(8))) u32 u32x8;

// Problem constants
constexpr int BB   = 2;
constexpr int SS   = 2048;
constexpr int HID  = 2048;
constexpr int NHD  = 16;
constexpr int DN   = 128;
constexpr int DR   = 64;
constexpr int DV   = 128;
constexpr int QLR  = 1536;
constexpr int KVLR = 512;
constexpr int QHD  = DN + DR;          // 192
constexpr int TT   = BB * SS;          // 4096 tokens

#ifndef USE_TDM
#define USE_TDM 1                      // Tensor Data Mover staging in attention
#endif

#if defined(__has_builtin)
# if __has_builtin(__builtin_amdgcn_s_wait_tensorcnt)
#  define WAIT_TENSORCNT0() __builtin_amdgcn_s_wait_tensorcnt(0)
# endif
#endif
#ifndef WAIT_TENSORCNT0
# define WAIT_TENSORCNT0() asm volatile("s_wait_tensorcnt 0x0" ::: "memory")
#endif

// ---------------------------------------------------------------------------
// TDM: issue a 2-D tile load (rows x rowElems bf16, contiguous rows) into LDS.
// D# per CDNA5 ISA ch.8: group0 = {flags, lds_addr, global_addr, type=2},
// group1 = {data_size=2B, tensor_dim0/1, tile_dim0/1, dim0_stride}.
// Wave-level op (EXEC ignored); tracked by TENSORcnt.
// ---------------------------------------------------------------------------
__device__ inline void tdm_load_tile(u32 ldsOff, const bf16_t* gsrc,
                                     u32 rowElems, u32 rows) {
  const unsigned long long ga = (unsigned long long)(uintptr_t)gsrc;
  u32x4 g0;
  g0[0] = 1u;                                           // count=1, user load
  g0[1] = ldsOff;                                       // LDS byte address
  g0[2] = (u32)ga;                                      // global_addr[31:0]
  g0[3] = ((u32)(ga >> 32) & 0x01FFFFFFu) | (2u << 30); // addr[56:32] | type=2
  u32x8 g1;
  g1[0] = 1u << 16;                                     // data_size = 2 bytes
  g1[1] = (rowElems & 0xFFFFu) << 16;                   // tensor_dim0 lo16
  g1[2] = (rowElems >> 16) | ((rows & 0xFFFFu) << 16);  // dim0 hi | dim1 lo
  g1[3] = ((rows >> 16) & 0xFFFFu) | ((rowElems & 0xFFFFu) << 16); // | tile_dim0
  g1[4] = rows & 0xFFFFu;                               // tile_dim1 (tile_dim2=0)
  g1[5] = rowElems;                                     // dim0_stride lo32
  g1[6] = 0u;                                           // stride hi / dim1_stride lo
  g1[7] = 0u;
  asm volatile("tensor_load_to_lds %0, %1" :: "s"(g0), "s"(g1) : "memory");
}

// ---------------------------------------------------------------------------
// f32 -> bf16 elementwise convert
// ---------------------------------------------------------------------------
__global__ __launch_bounds__(256)
void to_bf16(const float* __restrict__ x, bf16_t* __restrict__ y, long n) {
  long i = (long)blockIdx.x * 256 + threadIdx.x;
  long stride = (long)gridDim.x * 256;
  for (; i < n; i += stride) y[i] = (bf16_t)x[i];
}

// ---------------------------------------------------------------------------
// Tiled transpose + convert: W[K][N] f32 -> Wt[N][K] bf16 (coalesced both sides)
// ---------------------------------------------------------------------------
__global__ __launch_bounds__(256)
void transpose_to_bf16(const float* __restrict__ W, bf16_t* __restrict__ Wt,
                       int K, int N) {
  __shared__ float tile[32][33];
  const int kb = blockIdx.y * 32, nb = blockIdx.x * 32;
  const int tx = threadIdx.x & 31, ty = threadIdx.x >> 5;   // 32 x 8
#pragma unroll
  for (int i = ty; i < 32; i += 8) {
    int k = kb + i, n = nb + tx;
    tile[i][tx] = (k < K && n < N) ? W[(size_t)k * N + n] : 0.f;
  }
  __syncthreads();
#pragma unroll
  for (int i = ty; i < 32; i += 8) {
    int n = nb + i, k = kb + tx;
    if (n < N && k < K) Wt[(size_t)n * K + k] = (bf16_t)tile[tx][i];
  }
}

// ---------------------------------------------------------------------------
// TN WMMA GEMM: C[M][N](f32) = Act[M][K](bf16) x Wt[N][K](bf16)^T
// A-operand = Wt tile (16 out-cols, per-lane K-contiguous),
// B-operand = Act tile (16 act-rows/column-lanes, per-lane K-contiguous).
// Block 256 = 8 waves: 2 col-tiles x 4 row-groups -> block tile 256(M) x 32(N).
// All loads/stores are 16-32B contiguous per lane. EPI: 0 plain, 1 sigmoid.
// Requires: M % 256 == 0, N % 32 == 0, K % 32 == 0 (all shapes satisfy this).
// ---------------------------------------------------------------------------
template <int EPI>
__global__ __launch_bounds__(256)
void gemm_wmma_tn(const bf16_t* __restrict__ Act, const bf16_t* __restrict__ Wt,
                  float* __restrict__ C, int M, int N, int K) {
  const int lane = threadIdx.x & 31;
  const int wid  = threadIdx.x >> 5;
  const int wn   = wid & 1;
  const int wm   = wid >> 1;
  const int n0   = blockIdx.y * 32 + wn * 16;   // output-column base
  const int m0   = blockIdx.x * 256 + wm * 64;  // activation-row base
  const int l15  = lane & 15;
  const int half = lane >> 4;

  v8f zf = {};
  v8f acc[4];
#pragma unroll
  for (int t = 0; t < 4; ++t) acc[t] = zf;

  const bf16_t* wrow = Wt + (size_t)(n0 + l15) * K;

  for (int k0 = 0; k0 < K; k0 += 32) {
    if (k0 + 64 <= K) __builtin_prefetch(wrow + k0 + 32, 0, 0);  // global_prefetch_b8

    v16bf af;                                    // Wt fragment (A operand)
#pragma unroll
    for (int v = 0; v < 8; ++v) {
      const int kb = k0 + half * 16 + (v << 1);
      af[2 * v]     = wrow[kb];
      af[2 * v + 1] = wrow[kb + 1];
    }

#pragma unroll
    for (int t = 0; t < 4; ++t) {
      const bf16_t* arow = Act + (size_t)(m0 + t * 16 + l15) * K + k0 + half * 16;
      v16bf bfr;                                 // Act fragment (B operand)
#pragma unroll
      for (int e = 0; e < 16; ++e) bfr[e] = arow[e];
      acc[t] = __builtin_amdgcn_wmma_f32_16x16x32_bf16(
          false, af, false, bfr, (short)0, acc[t], false, false);
    }
  }

  // D: lane = act-row, VGPR r = out-col (r + 8*half) -> 8 contiguous floats/lane
#pragma unroll
  for (int t = 0; t < 4; ++t) {
    const int row = m0 + t * 16 + l15;
    float* cp = C + (size_t)row * N + n0 + half * 8;
#pragma unroll
    for (int r = 0; r < 8; ++r) {
      float x = acc[t][r];
      if (EPI == 1) x = 1.0f / (1.0f + __expf(-x));   // sigmoid gate
      cp[r] = x;
    }
  }
}

// ---------------------------------------------------------------------------
// RMS norm over L elements of each row (input row stride may exceed L), bf16 out
// ---------------------------------------------------------------------------
__global__ __launch_bounds__(256)
void rmsnorm_to_bf16(const float* __restrict__ x, const float* __restrict__ w,
                     bf16_t* __restrict__ y, int L, int strideIn) {
  const int row = blockIdx.x;
  const float* xr = x + (size_t)row * strideIn;
  float s = 0.f;
  for (int i = threadIdx.x; i < L; i += blockDim.x) {
    float v = xr[i];
    s += v * v;
  }
#pragma unroll
  for (int msk = 16; msk >= 1; msk >>= 1) s += __shfl_xor(s, msk, 32);
  __shared__ float red[8];
  if ((threadIdx.x & 31) == 0) red[threadIdx.x >> 5] = s;
  __syncthreads();
  float tot = 0.f;
#pragma unroll
  for (int i = 0; i < 8; ++i) tot += red[i];
  const float inv = rsqrtf(tot / (float)L + 1e-6f);
  for (int i = threadIdx.x; i < L; i += blockDim.x)
    y[(size_t)row * L + i] = (bf16_t)(xr[i] * inv * (1.0f + w[i]));
}

// ---------------------------------------------------------------------------
// RoPE (deinterleave + rotate-half) and repack to head-major bf16 Q/K/V
// ---------------------------------------------------------------------------
__global__ __launch_bounds__(256)
void rope_pack(const float* __restrict__ qup, const float* __restrict__ kvup,
               const float* __restrict__ ckv, const int* __restrict__ pos_ids,
               bf16_t* __restrict__ Qb, bf16_t* __restrict__ Kb,
               bf16_t* __restrict__ Vb) {
  const int t = blockIdx.x;            // token
  const int b = t / SS, s = t % SS;
  const int tid = threadIdx.x;

  __shared__ float cs[32], sn[32], kr[DR];
  if (tid < 32) {
    float p   = (float)pos_ids[s];
    float inv = __powf(10000.0f, -((float)tid) / 32.0f);   // base^(-2j/DR)
    float f   = p * inv;
    cs[tid] = __cosf(f);
    sn[tid] = __sinf(f);
  }
  __syncthreads();
  if (tid < DR) {
    const float* kro = ckv + (size_t)t * (KVLR + DR) + KVLR;
    const int i = tid;
    const float c = cs[i & 31], sv = sn[i & 31];
    float v;
    if (i < 32) v = kro[2 * i] * c - kro[2 * i + 1] * sv;
    else        v = kro[2 * (i - 32) + 1] * c + kro[2 * (i - 32)] * sv;
    kr[i] = v;
  }
  __syncthreads();

  for (int e = tid; e < NHD * QHD; e += 256) {
    const int h = e / QHD, d = e % QHD;
    const size_t dst = (((size_t)b * NHD + h) * SS + s) * QHD + d;
    const float* qrow = qup + (size_t)t * (NHD * QHD) + (size_t)h * QHD;
    float qv;
    if (d < DN) {
      qv = qrow[d];
    } else {
      const int i = d - DN;
      const float c = cs[i & 31], sv = sn[i & 31];
      const float* qr = qrow + DN;
      if (i < 32) qv = qr[2 * i] * c - qr[2 * i + 1] * sv;
      else        qv = qr[2 * (i - 32) + 1] * c + qr[2 * (i - 32)] * sv;
    }
    Qb[dst] = (bf16_t)qv;
    const float* krow = kvup + (size_t)t * (NHD * (DN + DV)) + (size_t)h * (DN + DV);
    Kb[dst] = (bf16_t)((d < DN) ? krow[d] : kr[d - DN]);
  }
  for (int e = tid; e < NHD * DV; e += 256) {
    const int h = e / DV, dv = e % DV;
    Vb[(((size_t)b * NHD + h) * SS + s) * DV + dv] =
        (bf16_t)kvup[(size_t)t * (NHD * (DN + DV)) + (size_t)h * (DN + DV) + DN + dv];
  }
}

// ---------------------------------------------------------------------------
// Flash attention with TDM-staged K/V tiles in LDS.
// grid (S/128, NH, B), block 256 (8 waves). Wave = 16 q rows; 32 keys/step.
// K/V tiles loaded ONCE per block (TDM or cooperative b128 fallback), all
// waves consume from LDS. Output fused with sigmoid gate -> bf16 [T, NH*DV].
// ---------------------------------------------------------------------------
__global__ __launch_bounds__(256)
void mla_flash_attn(const bf16_t* __restrict__ Qb, const bf16_t* __restrict__ Kb,
                    const bf16_t* __restrict__ Vb, const float* __restrict__ gate,
                    bf16_t* __restrict__ Og) {
  const int lane = threadIdx.x & 31;
  const int wid  = threadIdx.x >> 5;
  const int h    = blockIdx.y;
  const int b    = blockIdx.z;
  const int q0   = blockIdx.x * 128 + wid * 16;
  const int m    = lane & 15;
  const int half = lane >> 4;
  const size_t bh = (size_t)b * NHD + h;

  // Single shared block => LDS offsets are compile-time constants (base = 0)
  constexpr u32 K_ELEMS = 32 * QHD;          // 6144
  constexpr u32 V_ELEMS = 32 * DV;           // 4096
  __shared__ __align__(16) bf16_t smem[K_ELEMS + V_ELEMS + 8 * 16 * 32];
  bf16_t* Kl = smem;                         // LDS byte offset 0
  bf16_t* Vl = smem + K_ELEMS;               // LDS byte offset 12288
  bf16_t* Pw = smem + K_ELEMS + V_ELEMS + wid * (16 * 32);
  constexpr u32 K_OFF = 0;
  constexpr u32 V_OFF = K_ELEMS * 2;

  // Q fragments (16 x 192 = 6 chunks of K=32), per-lane contiguous loads
  const bf16_t* qp = Qb + (bh * SS + q0) * QHD;
  v16bf aq[6];
#pragma unroll
  for (int c = 0; c < 6; ++c) {
#pragma unroll
    for (int v = 0; v < 8; ++v) {
      const int kb = c * 32 + half * 16 + (v << 1);
      aq[c][2 * v]     = qp[(size_t)m * QHD + kb];
      aq[c][2 * v + 1] = qp[(size_t)m * QHD + kb + 1];
    }
  }

  v8f zf = {};
  v8f o[8];
#pragma unroll
  for (int j = 0; j < 8; ++j) o[j] = zf;
  float mrow[8], lrow[8];
#pragma unroll
  for (int r = 0; r < 8; ++r) { mrow[r] = -1e30f; lrow[r] = 0.f; }
  const float scale = 0.07216878364870323f;   // 1/sqrt(192)

  for (int k0 = 0; k0 < SS; k0 += 32) {
    // ---- stage K (32x192) and V (32x128) tiles into LDS, once per block
#if USE_TDM
    if (wid == 0) {
      tdm_load_tile(K_OFF, Kb + (bh * SS + k0) * QHD, QHD, 32);
      tdm_load_tile(V_OFF, Vb + (bh * SS + k0) * DV, DV, 32);
      WAIT_TENSORCNT0();
    }
#else
    {
      const int4* kg = (const int4*)(Kb + (bh * SS + k0) * QHD);
      const int4* vg = (const int4*)(Vb + (bh * SS + k0) * DV);
      int4* kl = (int4*)Kl;
      int4* vl = (int4*)Vl;
      const int tid = threadIdx.x;
#pragma unroll
      for (int c = 0; c < 3; ++c) kl[tid + 256 * c] = kg[tid + 256 * c];
#pragma unroll
      for (int c = 0; c < 2; ++c) vl[tid + 256 * c] = vg[tid + 256 * c];
    }
#endif
    __syncthreads();

    // ---- scores: Q(16x192) @ K^T -> two 16x16 tiles from LDS
    v8f s0 = zf, s1 = zf;
#pragma unroll
    for (int c = 0; c < 6; ++c) {
      v16bf b0, b1;
      const bf16_t* kp0 = Kl + (size_t)(lane & 15) * QHD + c * 32 + half * 16;
      const bf16_t* kp1 = kp0 + 16 * QHD;
#pragma unroll
      for (int e = 0; e < 16; ++e) { b0[e] = kp0[e]; b1[e] = kp1[e]; }
      s0 = __builtin_amdgcn_wmma_f32_16x16x32_bf16(false, aq[c], false, b0,
                                                   (short)0, s0, false, false);
      s1 = __builtin_amdgcn_wmma_f32_16x16x32_bf16(false, aq[c], false, b1,
                                                   (short)0, s1, false, false);
    }

    // ---- online softmax (row stats via 16-lane butterfly per half)
#pragma unroll
    for (int r = 0; r < 8; ++r) {
      float x0 = s0[r] * scale, x1 = s1[r] * scale;
      float tm = fmaxf(x0, x1);
#pragma unroll
      for (int msk = 8; msk >= 1; msk >>= 1) tm = fmaxf(tm, __shfl_xor(tm, msk, 32));
      const float nm = fmaxf(mrow[r], tm);
      const float p0 = __expf(x0 - nm);
      const float p1 = __expf(x1 - nm);
      float rs = p0 + p1;
#pragma unroll
      for (int msk = 8; msk >= 1; msk >>= 1) rs += __shfl_xor(rs, msk, 32);
      const float alpha = __expf(mrow[r] - nm);
      lrow[r] = lrow[r] * alpha + rs;
      mrow[r] = nm;
#pragma unroll
      for (int j = 0; j < 8; ++j) o[j][r] *= alpha;
      // P staging: same wave produces and consumes (LDS is in-order per wave)
      Pw[(r + half * 8) * 32 + (lane & 15)]      = (bf16_t)p0;
      Pw[(r + half * 8) * 32 + (lane & 15) + 16] = (bf16_t)p1;
    }

    // ---- P (16x32) @ V (32x128) from LDS
    v16bf pa;
#pragma unroll
    for (int v = 0; v < 8; ++v) {
      const int kb = half * 16 + (v << 1);
      pa[2 * v]     = Pw[m * 32 + kb];
      pa[2 * v + 1] = Pw[m * 32 + kb + 1];
    }
#pragma unroll
    for (int j = 0; j < 8; ++j) {
      v16bf bv;
#pragma unroll
      for (int v = 0; v < 8; ++v) {
#pragma unroll
        for (int jj = 0; jj < 2; ++jj) {
          const int kk = half * 16 + 2 * v + jj;
          bv[2 * v + jj] = Vl[(size_t)kk * DV + j * 16 + (lane & 15)];
        }
      }
      o[j] = __builtin_amdgcn_wmma_f32_16x16x32_bf16(false, pa, false, bv,
                                                     (short)0, o[j], false, false);
    }
    __syncthreads();   // all waves done with K/V tile before next stage
  }

  // ---- epilogue: 1/l normalization * sigmoid gate, token-major bf16 out
#pragma unroll
  for (int r = 0; r < 8; ++r) {
    const float invl = 1.0f / lrow[r];
    const int   qi   = q0 + r + half * 8;
    const size_t trow = ((size_t)b * SS + qi) * (size_t)(NHD * DV);
#pragma unroll
    for (int j = 0; j < 8; ++j) {
      const int col = h * DV + j * 16 + (lane & 15);
      Og[trow + col] = (bf16_t)(o[j][r] * invl * gate[trow + col]);
    }
  }
}

// ---------------------------------------------------------------------------
// Launcher
// ---------------------------------------------------------------------------
extern "C" void kernel_launch(void* const* d_in, const int* in_sizes, int n_in,
                              void* d_out, int out_size, void* d_ws, size_t ws_size,
                              hipStream_t stream) {
  (void)in_sizes; (void)n_in; (void)out_size; (void)ws_size;

  const float* hidden    = (const float*)d_in[0];
  const int*   pos_ids   = (const int*)d_in[1];
  const float* Wq_down   = (const float*)d_in[2];
  const float* q_norm_w  = (const float*)d_in[3];
  const float* Wq_up     = (const float*)d_in[4];
  const float* Wkv_down  = (const float*)d_in[5];
  const float* kv_norm_w = (const float*)d_in[6];
  const float* Wkv_up    = (const float*)d_in[7];
  const float* W_gate    = (const float*)d_in[8];
  const float* W_out     = (const float*)d_in[9];
  float* out = (float*)d_out;

  // Workspace carve-up
  char* ws = (char*)d_ws;
  float* qdown = (float*)ws;  ws += (size_t)TT * QLR * sizeof(float);
  float* ckv   = (float*)ws;  ws += (size_t)TT * (KVLR + DR) * sizeof(float);
  float* gateb = (float*)ws;  ws += (size_t)TT * (NHD * DV) * sizeof(float);
  float* qup   = (float*)ws;  ws += (size_t)TT * (NHD * QHD) * sizeof(float);
  float* kvup  = (float*)ws;  ws += (size_t)TT * (NHD * (DN + DV)) * sizeof(float);
  bf16_t* hbf   = (bf16_t*)ws; ws += (size_t)TT * HID * sizeof(bf16_t);
  bf16_t* qn    = (bf16_t*)ws; ws += (size_t)TT * QLR * sizeof(bf16_t);
  bf16_t* ckvn  = (bf16_t*)ws; ws += (size_t)TT * KVLR * sizeof(bf16_t);
  bf16_t* Qb    = (bf16_t*)ws; ws += (size_t)TT * (NHD * QHD) * sizeof(bf16_t);
  bf16_t* Kb    = (bf16_t*)ws; ws += (size_t)TT * (NHD * QHD) * sizeof(bf16_t);
  bf16_t* Vb    = (bf16_t*)ws; ws += (size_t)TT * (NHD * DV) * sizeof(bf16_t);
  bf16_t* attng = (bf16_t*)ws; ws += (size_t)TT * (NHD * DV) * sizeof(bf16_t);
  // transposed bf16 weights  Wt[N][K]
  bf16_t* WTqd  = (bf16_t*)ws; ws += (size_t)QLR * HID * sizeof(bf16_t);
  bf16_t* WTkvd = (bf16_t*)ws; ws += (size_t)(KVLR + DR) * HID * sizeof(bf16_t);
  bf16_t* WTg   = (bf16_t*)ws; ws += (size_t)(NHD * DV) * HID * sizeof(bf16_t);
  bf16_t* WTqu  = (bf16_t*)ws; ws += (size_t)(NHD * QHD) * QLR * sizeof(bf16_t);
  bf16_t* WTkvu = (bf16_t*)ws; ws += (size_t)(NHD * (DN + DV)) * KVLR * sizeof(bf16_t);
  bf16_t* WTo   = (bf16_t*)ws; ws += (size_t)HID * (NHD * DV) * sizeof(bf16_t);

  const dim3 blk(256);
  auto tgrid = [](int K, int N) { return dim3((N + 31) / 32, (K + 31) / 32); };
  auto ggrid = [](int M, int N) { return dim3(M / 256, N / 32); };

  // 0) precision conversion: hidden -> bf16; weights -> transposed bf16
  to_bf16<<<1024, blk, 0, stream>>>(hidden, hbf, (long)TT * HID);
  transpose_to_bf16<<<tgrid(HID, QLR), blk, 0, stream>>>(Wq_down, WTqd, HID, QLR);
  transpose_to_bf16<<<tgrid(HID, KVLR + DR), blk, 0, stream>>>(Wkv_down, WTkvd, HID, KVLR + DR);
  transpose_to_bf16<<<tgrid(HID, NHD * DV), blk, 0, stream>>>(W_gate, WTg, HID, NHD * DV);
  transpose_to_bf16<<<tgrid(QLR, NHD * QHD), blk, 0, stream>>>(Wq_up, WTqu, QLR, NHD * QHD);
  transpose_to_bf16<<<tgrid(KVLR, NHD * (DN + DV)), blk, 0, stream>>>(Wkv_up, WTkvu, KVLR, NHD * (DN + DV));
  transpose_to_bf16<<<tgrid(NHD * DV, HID), blk, 0, stream>>>(W_out, WTo, NHD * DV, HID);

  // 1) down projections + gate
  gemm_wmma_tn<0><<<ggrid(TT, QLR), blk, 0, stream>>>(hbf, WTqd, qdown, TT, QLR, HID);
  gemm_wmma_tn<0><<<ggrid(TT, KVLR + DR), blk, 0, stream>>>(hbf, WTkvd, ckv, TT, KVLR + DR, HID);
  gemm_wmma_tn<1><<<ggrid(TT, NHD * DV), blk, 0, stream>>>(hbf, WTg, gateb, TT, NHD * DV, HID);

  // 2) RMS norms -> bf16 latents
  rmsnorm_to_bf16<<<TT, blk, 0, stream>>>(qdown, q_norm_w, qn, QLR, QLR);
  rmsnorm_to_bf16<<<TT, blk, 0, stream>>>(ckv, kv_norm_w, ckvn, KVLR, KVLR + DR);

  // 3) up projections
  gemm_wmma_tn<0><<<ggrid(TT, NHD * QHD), blk, 0, stream>>>(qn, WTqu, qup, TT, NHD * QHD, QLR);
  gemm_wmma_tn<0><<<ggrid(TT, NHD * (DN + DV)), blk, 0, stream>>>(ckvn, WTkvu, kvup, TT, NHD * (DN + DV), KVLR);

  // 4) RoPE + repack to head-major bf16
  rope_pack<<<TT, blk, 0, stream>>>(qup, kvup, ckv, pos_ids, Qb, Kb, Vb);

  // 5) flash attention (TDM-staged K/V, fused gate epilogue)
  mla_flash_attn<<<dim3(SS / 128, NHD, BB), blk, 0, stream>>>(Qb, Kb, Vb, gateb, attng);

  // 6) output projection
  gemm_wmma_tn<0><<<ggrid(TT, HID), blk, 0, stream>>>(attng, WTo, out, TT, HID, NHD * DV);
}
